// Com_Out_Temp_model_12738873000455
// MI455X (gfx1250) — matrix operand order repaired and checked by reading, not compiled, not run on hardware
//
#include <hip/hip_runtime.h>
#include <hip/hip_bf16.h>

// ---------------------------------------------------------------------------
// CDNA5 (gfx1250) implementation of the batch-norm MLP model.
// Heavy layer (256->64 over 1M rows) runs on v_wmma_f32_16x16x32_bf16.
// Workspace use: ~46 KB (stats + folded weights + bf16 W1).
// B (=1M) is a multiple of 32; tiles are 32 rows.
// ---------------------------------------------------------------------------

typedef __attribute__((ext_vector_type(16))) __bf16 v16bf;
typedef __attribute__((ext_vector_type(8)))  float  v8f;

union Frag { v16bf v; unsigned int u[8]; };

__device__ __forceinline__ unsigned short f2bf(float f) {
  unsigned int u = __builtin_bit_cast(unsigned int, f);
  u += 0x7FFFu + ((u >> 16) & 1u);          // round-to-nearest-even
  return (unsigned short)(u >> 16);
}

__device__ __forceinline__ void features(const float* xr, float hi, float tph,
                                         float* f) {
  float lo = xr[5];
  float ratio = fminf(fmaxf(hi / lo, 3.0f), 100.0f);
  f[0] = hi;   f[1] = xr[2];        f[2] = xr[4];       f[3] = lo;
  f[4] = xr[3];f[5] = tph;          f[6] = ratio;       f[7] = xr[0] * xr[7];
  f[8] = xr[10] - hi;               f[9] = lo - xr[12];
}

// ---------------------------------------------------------------------------
// Pass 1: accumulate raw moments of the 10 features and of t = x[:,11].
// stat layout: [0..9]=sum f, [10..64]=sum f_k f_l (upper tri), [65]=sum t,
// [66]=sum t^2
// ---------------------------------------------------------------------------
__global__ void moments_kernel(const float* __restrict__ x,
                               const float* __restrict__ hip_,
                               const float* __restrict__ tph,
                               float* __restrict__ stat, int B) {
  float m[10]; float S[55]; float st = 0.f, st2 = 0.f;
#pragma unroll
  for (int k = 0; k < 10; k++) m[k] = 0.f;
#pragma unroll
  for (int i = 0; i < 55; i++) S[i] = 0.f;

  int stride = gridDim.x * blockDim.x;
  for (int r = blockIdx.x * blockDim.x + threadIdx.x; r < B; r += stride) {
    const float* xr = x + (size_t)r * 13;
    float f[10];
    features(xr, hip_[r], tph[r], f);
    int idx = 0;
#pragma unroll
    for (int k = 0; k < 10; k++) {
      m[k] += f[k];
#pragma unroll
      for (int l = k; l < 10; l++) { S[idx] += f[k] * f[l]; idx++; }
    }
    float t = xr[11];
    st += t; st2 += t * t;
  }

  __shared__ float red[67];
  for (int i = threadIdx.x; i < 67; i += blockDim.x) red[i] = 0.f;
  __syncthreads();
#pragma unroll
  for (int k = 0; k < 10; k++) atomicAdd(&red[k], m[k]);
#pragma unroll
  for (int i = 0; i < 55; i++) atomicAdd(&red[10 + i], S[i]);
  atomicAdd(&red[65], st);
  atomicAdd(&red[66], st2);
  __syncthreads();
  for (int i = threadIdx.x; i < 67; i += blockDim.x) atomicAdd(&stat[i], red[i]);
}

// ---------------------------------------------------------------------------
// Pass 2 (1 block): fold BN of layer 0 into effective weights, analytic stats.
// Also convert mlp2.w1 (64x256) to bf16 for WMMA.
// ---------------------------------------------------------------------------
__global__ void finalize0_kernel(const float* __restrict__ stat,
                                 const float* __restrict__ w0,   // 256x10
                                 const float* __restrict__ b0,   // 256
                                 const float* __restrict__ g0,
                                 const float* __restrict__ be0,
                                 const float* __restrict__ w1,   // 64x256
                                 const float* __restrict__ m1w0, // 8
                                 const float* __restrict__ m1b0,
                                 const float* __restrict__ m1g0,
                                 const float* __restrict__ m1be0,
                                 float invB,
                                 float* __restrict__ W0eff,      // 256x10
                                 float* __restrict__ beta0eff,   // 256
                                 unsigned short* __restrict__ w1bf, // 64x256
                                 float* __restrict__ m1l0) {     // 16
  __shared__ float m[10], S[10][10], Et, Et2;
  if (threadIdx.x == 0) {
    for (int k = 0; k < 10; k++) m[k] = stat[k] * invB;
    int idx = 10;
    for (int k = 0; k < 10; k++)
      for (int l = k; l < 10; l++) {
        float v = stat[idx++] * invB;
        S[k][l] = v; S[l][k] = v;
      }
    Et = stat[65] * invB;
    Et2 = stat[66] * invB;
  }
  __syncthreads();

  for (int j = threadIdx.x; j < 256; j += blockDim.x) {
    const float* wj = w0 + j * 10;
    float wm = 0.f;
#pragma unroll
    for (int k = 0; k < 10; k++) wm += wj[k] * m[k];
    float mu = wm + b0[j];
    float q = 0.f;
#pragma unroll
    for (int k = 0; k < 10; k++) {
      float a = 0.f;
#pragma unroll
      for (int l = 0; l < 10; l++) a += S[k][l] * wj[l];
      q += wj[k] * a;
    }
    float Eh2 = q + 2.f * b0[j] * wm + b0[j] * b0[j];
    float var = Eh2 - mu * mu;
    float alpha = g0[j] * rsqrtf(var + 1e-5f);
    float beta = be0[j] - mu * alpha;
#pragma unroll
    for (int k = 0; k < 10; k++) W0eff[j * 10 + k] = wj[k] * alpha;
    beta0eff[j] = b0[j] * alpha + beta;
  }

  for (int i = threadIdx.x; i < 64 * 256; i += blockDim.x)
    w1bf[i] = f2bf(w1[i]);

  if (threadIdx.x < 8) {
    int j = threadIdx.x;
    float w = m1w0[j];
    float mu = w * Et + m1b0[j];
    float var = w * w * (Et2 - Et * Et);
    float alpha = m1g0[j] * rsqrtf(var + 1e-5f);
    float beta = m1be0[j] - mu * alpha;
    m1l0[j] = w * alpha;
    m1l0[8 + j] = m1b0[j] * alpha + beta;
  }
}

// ---------------------------------------------------------------------------
// Pass 4 (1 block): BN stats -> alpha/beta for layer 1 of both MLPs.
// stats2: [0..63]=sum h2, [64..127]=sum h2^2, [128..135]=mlp1 sum,
//         [136..143]=mlp1 sum^2
// ---------------------------------------------------------------------------
__global__ void finalize1_kernel(const float* __restrict__ stats2,
                                 const float* __restrict__ g1,
                                 const float* __restrict__ be1,
                                 const float* __restrict__ m1g1,
                                 const float* __restrict__ m1be1,
                                 float invB,
                                 float* __restrict__ bn1,    // 128
                                 float* __restrict__ m1bn1)  // 16
{
  int i = threadIdx.x;
  if (i < 64) {
    float mu = stats2[i] * invB;
    float var = stats2[64 + i] * invB - mu * mu;
    float a = g1[i] * rsqrtf(var + 1e-5f);
    bn1[i] = a;
    bn1[64 + i] = be1[i] - mu * a;
  } else if (i < 72) {
    int j = i - 64;
    float mu = stats2[128 + j] * invB;
    float var = stats2[136 + j] * invB - mu * mu;
    float a = m1g1[j] * rsqrtf(var + 1e-5f);
    m1bn1[j] = a;
    m1bn1[8 + j] = m1be1[j] - mu * a;
  }
}

// ---------------------------------------------------------------------------
// Main kernel. PASS=0: accumulate BN1 statistics.  PASS=1: produce output.
// Block = 256 threads (8 wave32).  Tile = 32 rows.
// Wave w: N-tile = w&3 (cols 16*(w&3)..), row half = w>>2 (rows 0-15 / 16-31).
// a1 tile staged in LDS as bf16; W1 fragments preloaded in registers.
// ---------------------------------------------------------------------------
template <int PASS>
__global__ __launch_bounds__(256) void
mlp_main_kernel(const float* __restrict__ x, const float* __restrict__ hip_,
                const float* __restrict__ tph,
                const float* __restrict__ W0eff,
                const float* __restrict__ beta0eff,
                const unsigned short* __restrict__ w1bf,
                const float* __restrict__ b1, float* __restrict__ stats2,
                const float* __restrict__ bn1, const float* __restrict__ w2,
                const float* __restrict__ b2, const float* __restrict__ m1l0,
                const float* __restrict__ m1w1, const float* __restrict__ m1b1,
                const float* __restrict__ m1bn1, const float* __restrict__ m1w2,
                const float* __restrict__ m1b2, float* __restrict__ out, int B,
                int numTiles) {
  __shared__ float sW0[256 * 10];
  __shared__ float sBeta0[256];
  __shared__ unsigned short sA1[32 * 256];   // bf16 a1 tile, row-major
  __shared__ float sF[32][10];
  __shared__ float sB1[64], sW2v[64], sAl1[64], sBe1v[64];
  __shared__ float sColSum[64], sColSq[64];
  __shared__ float sOut[32];
  __shared__ float sM1l0[16], sM1w1[64], sM1b1[8], sM1bn[16], sM1w2[8];
  __shared__ float sM1Stat[16];
  __shared__ float sScal[2];                 // [0]=mlp2.b2, [1]=mlp1.b2

  const int t = threadIdx.x;
  const int lane = t & 31, wv = t >> 5;
  const int ntile = wv & 3, rhalf = wv >> 2;
  const int lh = lane >> 4;
  const int n = ntile * 16 + (lane & 15);

  // ---- stage block-wide constants ----
  for (int i = t; i < 2560; i += 256) sW0[i] = W0eff[i];
  if (t < 256) sBeta0[t] = beta0eff[t];
  if (t < 64) {
    sB1[t] = b1[t];
    sW2v[t] = w2[t];
    sAl1[t] = (PASS == 1) ? bn1[t] : 0.f;
    sBe1v[t] = (PASS == 1) ? bn1[64 + t] : 0.f;
    sColSum[t] = 0.f;
    sColSq[t] = 0.f;
    sM1w1[t] = m1w1[t];
  }
  if (t < 16) {
    sM1l0[t] = m1l0[t];
    sM1Stat[t] = 0.f;
    sM1bn[t] = (PASS == 1) ? m1bn1[t] : 0.f;
  }
  if (t < 8) { sM1b1[t] = m1b1[t]; sM1w2[t] = m1w2[t]; }
  if (t == 0) { sScal[0] = b2[0]; sScal[1] = m1b2[0]; }

  // ---- preload B (W1) fragments into registers ----
  // B fragment, contiguous k order used here: lane-half lh holds K = 16*lh + {2v,2v+1} per VGPR v
  Frag bfrag[8];
  const unsigned int* w1u = (const unsigned int*)w1bf;
#pragma unroll
  for (int ks = 0; ks < 8; ks++) {
#pragma unroll
    for (int v = 0; v < 8; v++) {
      int k0 = ks * 32 + lh * 16 + 2 * v;
      bfrag[ks].u[v] = w1u[n * 128 + (k0 >> 1)];
    }
  }
  __syncthreads();

  float m1sum[8], m1sq[8];
#pragma unroll
  for (int i = 0; i < 8; i++) { m1sum[i] = 0.f; m1sq[i] = 0.f; }

  for (int tile = blockIdx.x; tile < numTiles; tile += gridDim.x) {
    const int rowBase = tile * 32;

    // ---- phase A: features + mlp1 scalar chain (row leaders) ----
    if (t < 32) {
      int row = rowBase + t;
      const float* xr = x + (size_t)row * 13;
      float f[10];
      features(xr, hip_[row], tph[row], f);
#pragma unroll
      for (int k = 0; k < 10; k++) sF[t][k] = f[k];

      float t11 = xr[11];
      float a1p[8], h2p[8];
#pragma unroll
      for (int i = 0; i < 8; i++)
        a1p[i] = tanhf(sM1l0[i] * t11 + sM1l0[8 + i]);
#pragma unroll
      for (int i = 0; i < 8; i++) {
        float s = sM1b1[i];
#pragma unroll
        for (int j = 0; j < 8; j++) s += sM1w1[i * 8 + j] * a1p[j];
        h2p[i] = s;
      }
      if (PASS == 0) {
#pragma unroll
        for (int i = 0; i < 8; i++) {
          m1sum[i] += h2p[i];
          m1sq[i] += h2p[i] * h2p[i];
        }
      } else {
        float o1 = sScal[1];
#pragma unroll
        for (int i = 0; i < 8; i++)
          o1 += sM1w2[i] * tanhf(sM1bn[i] * h2p[i] + sM1bn[8 + i]);
        sOut[t] = o1 + sScal[0];
      }
    }
    __syncthreads();

    // ---- phase B: a1 = tanh(W0eff f + beta0eff) -> LDS (bf16) ----
    {
      int rl = t & 31, jb = (t >> 5) * 32;
      float f[10];
#pragma unroll
      for (int k = 0; k < 10; k++) f[k] = sF[rl][k];
      unsigned short* arow = sA1 + rl * 256;
      for (int j = jb; j < jb + 32; j++) {
        const float* wj = sW0 + j * 10;
        float s = sBeta0[j];
#pragma unroll
        for (int k = 0; k < 10; k++) s += wj[k] * f[k];
        arow[j] = f2bf(tanhf(s));
      }
    }
    __syncthreads();

    // ---- phase C: WMMA 16x16 tile over K=256 (8 x K=32) ----
    const int mloc = rhalf * 16 + (lane & 15);
    v8f acc = {0.f, 0.f, 0.f, 0.f, 0.f, 0.f, 0.f, 0.f};
    const unsigned int* arow = (const unsigned int*)(sA1 + mloc * 256);
#pragma unroll
    for (int ks = 0; ks < 8; ks++) {
      Frag a;
      // A fragment, same contiguous k order as B: K = 16*lh + 2v
#pragma unroll
      for (int v = 0; v < 8; v++) {
        int k0 = ks * 32 + 16 * lh + 2 * v;
        a.u[v] = arow[k0 >> 1];
      }
      acc = __builtin_amdgcn_wmma_f32_16x16x32_bf16(
          false, a.v, false, bfrag[ks].v, (short)0, acc, false, false);
    }

    float b1n = sB1[n];
    if (PASS == 0) {
      float s = 0.f, ss = 0.f;
#pragma unroll
      for (int i = 0; i < 8; i++) {
        float v = acc[i] + b1n;
        s += v; ss += v * v;
      }
      atomicAdd(&sColSum[n], s);
      atomicAdd(&sColSq[n], ss);
    } else {
      float al = sAl1[n], be = sBe1v[n], wn = sW2v[n];
#pragma unroll
      for (int i = 0; i < 8; i++) {
        int rloc = rhalf * 16 + (lh ? 8 + i : i);
        atomicAdd(&sOut[rloc], wn * tanhf(al * (acc[i] + b1n) + be));
      }
    }
    __syncthreads();

    // ---- phase D: write output rows ----
    if (PASS == 1 && t < 32) out[rowBase + t] = sOut[t];
    __syncthreads();
  }

  // ---- final reduction of BN1 statistics ----
  if (PASS == 0) {
    if (t < 32) {
#pragma unroll
      for (int i = 0; i < 8; i++) {
        atomicAdd(&sM1Stat[i], m1sum[i]);
        atomicAdd(&sM1Stat[8 + i], m1sq[i]);
      }
    }
    __syncthreads();
    if (t < 64) {
      atomicAdd(&stats2[t], sColSum[t]);
      atomicAdd(&stats2[64 + t], sColSq[t]);
    }
    if (t < 16) atomicAdd(&stats2[128 + t], sM1Stat[t]);
  }
}

// ---------------------------------------------------------------------------
// Host launcher.
// d_in order (jax tree-flatten, dict keys alphabetical inside 'params'):
// 0:x 1:hi_pressure 2:temp_p_h_5
// 3..12  mlp1: b0 b1 b2 be0 be1 g0 g1 w0 w1 w2
// 13..22 mlp2: b0 b1 b2 be0 be1 g0 g1 w0 w1 w2
// ---------------------------------------------------------------------------
extern "C" void kernel_launch(void* const* d_in, const int* in_sizes, int n_in,
                              void* d_out, int out_size, void* d_ws,
                              size_t ws_size, hipStream_t stream) {
  const float* x    = (const float*)d_in[0];
  const float* hip_ = (const float*)d_in[1];
  const float* tph  = (const float*)d_in[2];
  const float* m1b0  = (const float*)d_in[3];
  const float* m1b1  = (const float*)d_in[4];
  const float* m1b2  = (const float*)d_in[5];
  const float* m1be0 = (const float*)d_in[6];
  const float* m1be1 = (const float*)d_in[7];
  const float* m1g0  = (const float*)d_in[8];
  const float* m1g1  = (const float*)d_in[9];
  const float* m1w0  = (const float*)d_in[10];
  const float* m1w1  = (const float*)d_in[11];
  const float* m1w2  = (const float*)d_in[12];
  const float* b0  = (const float*)d_in[13];
  const float* b1  = (const float*)d_in[14];
  const float* b2  = (const float*)d_in[15];
  const float* be0 = (const float*)d_in[16];
  const float* be1 = (const float*)d_in[17];
  const float* g0  = (const float*)d_in[18];
  const float* g1  = (const float*)d_in[19];
  const float* w0  = (const float*)d_in[20];
  const float* w1  = (const float*)d_in[21];
  const float* w2  = (const float*)d_in[22];

  const int B = in_sizes[0] / 13;
  const float invB = 1.0f / (float)B;

  // Workspace layout (floats)
  float* ws = (float*)d_ws;
  float* stat1    = ws;           // 67
  float* stats2   = ws + 67;      // 144
  float* W0eff    = ws + 256;     // 2560
  float* beta0eff = ws + 2816;    // 256
  float* m1l0     = ws + 3072;    // 16
  float* bn1      = ws + 3088;    // 128
  float* m1bn1    = ws + 3216;    // 16
  unsigned short* w1bf = (unsigned short*)(ws + 3232); // 16384 u16

  hipMemsetAsync(ws, 0, 211 * sizeof(float), stream);

  moments_kernel<<<512, 256, 0, stream>>>(x, hip_, tph, stat1, B);

  finalize0_kernel<<<1, 256, 0, stream>>>(stat1, w0, b0, g0, be0, w1, m1w0,
                                          m1b0, m1g0, m1be0, invB, W0eff,
                                          beta0eff, w1bf, m1l0);

  const int numTiles = B / 32;  // B is a multiple of 32
  int grid = numTiles < 2048 ? numTiles : 2048;

  mlp_main_kernel<0><<<grid, 256, 0, stream>>>(
      x, hip_, tph, W0eff, beta0eff, w1bf, b1, stats2, bn1, w2, b2, m1l0,
      m1w1, m1b1, m1bn1, m1w2, m1b2, (float*)d_out, B, numTiles);

  finalize1_kernel<<<1, 128, 0, stream>>>(stats2, g1, be1, m1g1, m1be1, invB,
                                          bn1, m1bn1);

  mlp_main_kernel<1><<<grid, 256, 0, stream>>>(
      x, hip_, tph, W0eff, beta0eff, w1bf, b1, stats2, bn1, w2, b2, m1l0,
      m1w1, m1b1, m1bn1, m1w2, m1b2, (float*)d_out, B, numTiles);
}
